// MeshDiT_14474039787747
// MI455X (gfx1250) — 8-bit matrix operand order repaired and checked by reading, compiled, not run on hardware
//
#include <hip/hip_runtime.h>
#include <stdint.h>

#define H_     1024
#define NH_    16
#define DH_    64
#define MLP_   4096
#define LS_    512
#define LC_    128
#define TT_    640
#define IN_    64
#define FREQ_  256
#define EPSF   1e-6f

typedef __attribute__((ext_vector_type(8))) int v8i;
typedef __attribute__((ext_vector_type(4))) int v4i;
typedef long long i64;

__global__ __launch_bounds__(128) void gemm_iu8(
    const int8_t* __restrict__ A, const float* __restrict__ ascale,
    const int8_t* __restrict__ B, const float* __restrict__ wscale,
    const float* __restrict__ bias, float* __restrict__ C,
    int M, int N, int K, int ldc)
{
    const int lane = threadIdx.x & 31;
    const int wave = threadIdx.x >> 5;
    const int m0   = blockIdx.y * 64 + (wave >> 1) * 32;
    const int n0   = blockIdx.x * 64 + (wave & 1) * 32;
    const int l15  = lane & 15;
    const int hi   = (lane >> 4) & 1;

    const int8_t* Ap0 = A + (size_t)(m0 + l15) * K + hi * 16;
    const int8_t* Ap1 = Ap0 + (size_t)16 * K;
    const int8_t* Bp0 = B + (size_t)(n0 + l15) * K + hi * 16;
    const int8_t* Bp1 = Bp0 + (size_t)16 * K;

    v8i acc00 = {0,0,0,0,0,0,0,0};
    v8i acc01 = {0,0,0,0,0,0,0,0};
    v8i acc10 = {0,0,0,0,0,0,0,0};
    v8i acc11 = {0,0,0,0,0,0,0,0};

    for (int k0 = 0; k0 < K; k0 += 64) {
        union { v8i v; i64 d[4]; } a0, a1;
        const i64* ap0 = (const i64*)(Ap0 + k0);
        const i64* ap1 = (const i64*)(Ap1 + k0);
        a0.d[0] = ap0[0]; a0.d[1] = ap0[1]; a0.d[2] = ap0[4]; a0.d[3] = ap0[5];
        a1.d[0] = ap1[0]; a1.d[1] = ap1[1]; a1.d[2] = ap1[4]; a1.d[3] = ap1[5];

        union { v8i v; v4i q[2]; } b0, b1;
        const v4i* bp0 = (const v4i*)(Bp0 + k0);
        const v4i* bp1 = (const v4i*)(Bp1 + k0);
        b0.q[0] = bp0[0]; b0.q[1] = bp0[2];
        b1.q[0] = bp1[0]; b1.q[1] = bp1[2];

        __builtin_prefetch(Bp0 + k0 + 256, 0, 1);
        __builtin_prefetch(Bp1 + k0 + 256, 0, 1);

        acc00 = __builtin_amdgcn_wmma_i32_16x16x64_iu8(true, a0.v, true, b0.v, acc00, false, false);
        acc01 = __builtin_amdgcn_wmma_i32_16x16x64_iu8(true, a0.v, true, b1.v, acc01, false, false);
        acc10 = __builtin_amdgcn_wmma_i32_16x16x64_iu8(true, a1.v, true, b0.v, acc10, false, false);
        acc11 = __builtin_amdgcn_wmma_i32_16x16x64_iu8(true, a1.v, true, b1.v, acc11, false, false);
    }

    const float ws = wscale[0];
    const int na = n0 + l15;
    const int nb = na + 16;
    const float bna = bias ? bias[na] : 0.0f;
    const float bnb = bias ? bias[nb] : 0.0f;
#pragma unroll
    for (int r = 0; r < 8; ++r) {
        const int ma = m0 + hi * 8 + r;
        const int mb = ma + 16;
        const float sa = ascale[ma] * ws;
        const float sb = ascale[mb] * ws;
        C[(size_t)ma * ldc + na] = (float)acc00[r] * sa + bna;
        C[(size_t)ma * ldc + nb] = (float)acc01[r] * sa + bnb;
        C[(size_t)mb * ldc + na] = (float)acc10[r] * sb + bna;
        C[(size_t)mb * ldc + nb] = (float)acc11[r] * sb + bnb;
    }
}

__global__ void k_act_quant(const float* __restrict__ X, int K,
                            int8_t* __restrict__ Q, float* __restrict__ ascale)
{
    const int row = blockIdx.x;
    const float* x = X + (size_t)row * K;
    int8_t*      q = Q + (size_t)row * K;
    __shared__ float rs[256], rm[256];
    float ss = 0.f, mm = 0.f;
    for (int i = threadIdx.x; i < K; i += 256) {
        float v = x[i]; ss += v * v; mm = fmaxf(mm, fabsf(v));
    }
    rs[threadIdx.x] = ss; rm[threadIdx.x] = mm; __syncthreads();
    for (int st = 128; st > 0; st >>= 1) {
        if (threadIdx.x < st) {
            rs[threadIdx.x] += rs[threadIdx.x + st];
            rm[threadIdx.x]  = fmaxf(rm[threadIdx.x], rm[threadIdx.x + st]);
        }
        __syncthreads();
    }
    const float r  = rsqrtf(rs[0] / (float)K + EPSF);
    const float mx = fmaxf(rm[0] * r, 1e-5f);
    const float s  = 127.0f / mx;
    if (threadIdx.x == 0) ascale[row] = mx * (1.0f / 127.0f);
    for (int i = threadIdx.x; i < K; i += 256) {
        float v = rintf(x[i] * r * s);
        v = fminf(fmaxf(v, -128.f), 127.f);
        q[i] = (int8_t)(int)v;
    }
}

__global__ void k_wabs_part(const float* __restrict__ W, size_t n, float* __restrict__ part)
{
    __shared__ float r[256];
    float s = 0.f;
    for (size_t i = (size_t)blockIdx.x * 256 + threadIdx.x; i < n; i += (size_t)gridDim.x * 256)
        s += fabsf(W[i]);
    r[threadIdx.x] = s; __syncthreads();
    for (int st = 128; st > 0; st >>= 1) {
        if (threadIdx.x < st) r[threadIdx.x] += r[threadIdx.x + st];
        __syncthreads();
    }
    if (threadIdx.x == 0) part[blockIdx.x] = r[0];
}
__global__ void k_wabs_fin(const float* __restrict__ part, int np, float invn,
                           float* __restrict__ scale)
{
    __shared__ float r[256];
    float s = 0.f;
    for (int i = threadIdx.x; i < np; i += 256) s += part[i];
    r[threadIdx.x] = s; __syncthreads();
    for (int st = 128; st > 0; st >>= 1) {
        if (threadIdx.x < st) r[threadIdx.x] += r[threadIdx.x + st];
        __syncthreads();
    }
    if (threadIdx.x == 0) scale[0] = fmaxf(r[0] * invn, 1e-5f);
}
__global__ void k_wquant(const float* __restrict__ W, const float* __restrict__ scale,
                         int8_t* __restrict__ Q, size_t n)
{
    size_t i = (size_t)blockIdx.x * 256 + threadIdx.x;
    if (i >= n) return;
    float ws = 1.0f / scale[0];
    float q  = rintf(W[i] * ws);
    q = fminf(fmaxf(q, -1.f), 1.f);
    Q[i] = (int8_t)(int)q;
}

__global__ void k_ln_mod(const float* __restrict__ X, const float* __restrict__ scale,
                         const float* __restrict__ shift, float* __restrict__ out)
{
    const int row = blockIdx.x;
    const float* x = X + (size_t)row * H_;
    float*       o = out + (size_t)row * H_;
    __shared__ float rs[256], rq[256];
    float s = 0.f, q = 0.f;
    for (int i = threadIdx.x; i < H_; i += 256) { float v = x[i]; s += v; q += v * v; }
    rs[threadIdx.x] = s; rq[threadIdx.x] = q; __syncthreads();
    for (int st = 128; st > 0; st >>= 1) {
        if (threadIdx.x < st) { rs[threadIdx.x] += rs[threadIdx.x + st]; rq[threadIdx.x] += rq[threadIdx.x + st]; }
        __syncthreads();
    }
    const float mean = rs[0] * (1.0f / H_);
    const float var  = rq[0] * (1.0f / H_) - mean * mean;
    const float r    = rsqrtf(var + EPSF);
    for (int i = threadIdx.x; i < H_; i += 256) {
        float y = (x[i] - mean) * r;
        float m = 1.0f + scale[i];
        float sh = shift ? shift[i] : 0.0f;
        o[i] = y * m + sh;
    }
}

__global__ __launch_bounds__(256) void k_hgrn_scan(
    const float* __restrict__ q, const float* __restrict__ f,
    const float* __restrict__ v, float* __restrict__ o, int T)
{
    const int head = blockIdx.x;
    const int tid  = threadIdx.x;
    const int vi   = tid & 63;
    const int kc   = tid >> 6;
    __shared__ float qs[64], fs[64], vs[64], ps[4][64];
    float S[16];
#pragma unroll
    for (int i = 0; i < 16; ++i) S[i] = 0.f;
    const size_t hoff = (size_t)head * DH_;
    for (int t = 0; t < T; ++t) {
        const size_t base = (size_t)t * H_ + hoff;
        if (tid < 64)        qs[tid]       = q[base + tid];
        else if (tid < 128)  fs[tid - 64]  = f[base + tid - 64];
        else if (tid < 192)  vs[tid - 128] = v[base + tid - 128];
        __syncthreads();
        const float vv = vs[vi];
        float acc = 0.f;
#pragma unroll
        for (int j = 0; j < 16; ++j) {
            const int k = kc * 16 + j;
            const float fk = fs[k];
            S[j] = fk * S[j] + (1.0f - fk) * vv;
            acc += qs[k] * S[j];
        }
        ps[kc][vi] = acc;
        __syncthreads();
        if (kc == 0) o[base + vi] = ps[0][vi] + ps[1][vi] + ps[2][vi] + ps[3][vi];
        __syncthreads();
    }
}

__global__ void k_gnorm_gate(float* __restrict__ o, const float* __restrict__ g,
                             const float* __restrict__ gnorm)
{
    const int b = blockIdx.x, d = threadIdx.x;
    const size_t base = (size_t)b * DH_;
    __shared__ float r[64];
    const float vv = o[base + d];
    r[d] = vv * vv; __syncthreads();
    for (int st = 32; st > 0; st >>= 1) { if (d < st) r[d] += r[d + st]; __syncthreads(); }
    const float rr = rsqrtf(r[0] * (1.0f / DH_) + EPSF);
    const float gg = g[base + d];
    o[base + d] = vv * rr * gnorm[d] * gg / (1.0f + expf(-gg));
}

__global__ void k_sigmoid(float* x, int n)
{ int i = blockIdx.x * 256 + threadIdx.x; if (i < n) x[i] = 1.0f / (1.0f + expf(-x[i])); }

__global__ void k_silu(const float* __restrict__ x, float* __restrict__ y, int n)
{ int i = blockIdx.x * 256 + threadIdx.x; if (i < n) { float v = x[i]; y[i] = v / (1.0f + expf(-v)); } }

__global__ void k_copy(float* __restrict__ d, const float* __restrict__ s, int n)
{ int i = blockIdx.x * 256 + threadIdx.x; if (i < n) d[i] = s[i]; }

__global__ void k_silugate(const float* __restrict__ big, float* __restrict__ mid, int n)
{
    int i = blockIdx.x * 256 + threadIdx.x; if (i >= n) return;
    int m = i >> 12, j = i & 4095;
    float gv = big[(size_t)m * 8192 + j];
    float yv = big[(size_t)m * 8192 + 4096 + j];
    mid[i] = (gv / (1.0f + expf(-gv))) * yv;
}

__global__ void k_gated_add(float* __restrict__ dst, const float* __restrict__ src,
                            const float* __restrict__ gate, int n)
{
    int i = blockIdx.x * 256 + threadIdx.x; if (i >= n) return;
    int j = i & (H_ - 1);
    dst[i] += gate[j] * src[i];
}

__global__ void k_matvec(const float* __restrict__ W, const float* __restrict__ bias,
                         const float* __restrict__ x, float* __restrict__ out,
                         int N, int K, int silu_out)
{
    int n = blockIdx.x * 256 + threadIdx.x; if (n >= N) return;
    const float* w = W + (size_t)n * K;
    float acc = bias ? bias[n] : 0.0f;
    for (int k = 0; k < K; ++k) acc += x[k] * w[k];
    if (silu_out) acc = acc / (1.0f + expf(-acc));
    out[n] = acc;
}

__global__ void k_timestep_emb(const float* __restrict__ t, float* __restrict__ emb)
{
    int i = threadIdx.x;
    float fr = expf(-logf(10000.0f) * (float)i / 128.0f);
    float a  = t[0] * fr;
    emb[i] = cosf(a); emb[i + 128] = sinf(a);
}

extern "C" void kernel_launch(void* const* d_in, const int* in_sizes, int n_in,
                              void* d_out, int out_size, void* d_ws, size_t ws_size,
                              hipStream_t stream)
{
    (void)in_sizes; (void)n_in; (void)out_size; (void)ws_size;

    int ii = 0;
    auto F = [&]() -> const float* { return (const float*)d_in[ii++]; };

    const float* x       = F();
    const float* context = F();
    const float* t       = F();
    const float* xemb_w  = F(); const float* xemb_b = F();
    const float* te_w1 = F(); const float* te_b1 = F();
    const float* te_w2 = F(); const float* te_b2 = F();

    struct DualW { const float *ada_x_w,*ada_x_b,*ada_c_w,*ada_c_b,*to_q,*add_q,*to_out,*to_add_out,
                   *wq,*wf,*wi,*wg,*wo,*gnorm,*ffg,*ffd,*ffcg,*ffcd; } du[2];
    for (int d = 0; d < 2; ++d) {
        du[d].ada_x_w = F(); du[d].ada_x_b = F(); du[d].ada_c_w = F(); du[d].ada_c_b = F();
        du[d].to_q = F(); du[d].add_q = F(); du[d].to_out = F(); du[d].to_add_out = F();
        du[d].wq = F(); du[d].wf = F(); du[d].wi = F(); du[d].wg = F(); du[d].wo = F();
        du[d].gnorm = F();
        du[d].ffg = F(); du[d].ffd = F(); du[d].ffcg = F(); du[d].ffcd = F();
    }
    struct SingleW { const float *ada_w,*ada_b,*wq,*wf,*wi,*wg,*wo,*gnorm,*mg,*md,*pw,*pb; } si[2];
    for (int s = 0; s < 2; ++s) {
        si[s].ada_w = F(); si[s].ada_b = F();
        si[s].wq = F(); si[s].wf = F(); si[s].wi = F(); si[s].wg = F(); si[s].wo = F();
        si[s].gnorm = F();
        si[s].mg = F(); si[s].md = F();
        si[s].pw = F(); si[s].pb = F();
    }
    const float* fada_w = F(); const float* fada_b = F();
    const float* flin_w = F(); const float* flin_b = F();

    char* wsbase = (char*)d_ws; size_t off = 0;
    auto alloc = [&](size_t nbytes) -> void* {
        size_t o = (off + 255) & ~(size_t)255; off = o + nbytes; return wsbase + o;
    };

    struct QW { int8_t* q; float* scale; int N, K; };
    float* wpart = (float*)alloc(256 * sizeof(float));
    auto prep = [&](const float* w, int N, int K) -> QW {
        QW r; r.N = N; r.K = K;
        r.q = (int8_t*)alloc((size_t)N * K);
        r.scale = (float*)alloc(sizeof(float));
        size_t n = (size_t)N * K;
        k_wabs_part<<<256, 256, 0, stream>>>(w, n, wpart);
        k_wabs_fin<<<1, 256, 0, stream>>>(wpart, 256, 1.0f / (float)n, r.scale);
        k_wquant<<<(unsigned)((n + 255) / 256), 256, 0, stream>>>(w, r.scale, r.q, n);
        return r;
    };

    QW qxemb = prep(xemb_w, H_, IN_);
    struct DQ { QW to_q, add_q, to_out, to_add_out, wq, wf, wi, wg, wo, ffg, ffd, ffcg, ffcd; } dq[2];
    for (int d = 0; d < 2; ++d) {
        dq[d].to_q = prep(du[d].to_q, H_, H_);   dq[d].add_q = prep(du[d].add_q, H_, H_);
        dq[d].to_out = prep(du[d].to_out, H_, H_); dq[d].to_add_out = prep(du[d].to_add_out, H_, H_);
        dq[d].wq = prep(du[d].wq, H_, H_); dq[d].wf = prep(du[d].wf, H_, H_);
        dq[d].wi = prep(du[d].wi, H_, H_); dq[d].wg = prep(du[d].wg, H_, H_);
        dq[d].wo = prep(du[d].wo, H_, H_);
        dq[d].ffg = prep(du[d].ffg, 2 * MLP_, H_); dq[d].ffd = prep(du[d].ffd, H_, MLP_);
        dq[d].ffcg = prep(du[d].ffcg, 2 * MLP_, H_); dq[d].ffcd = prep(du[d].ffcd, H_, MLP_);
    }
    struct SQ { QW wq, wf, wi, wg, wo, mg, md, pw; } sq[2];
    for (int s = 0; s < 2; ++s) {
        sq[s].wq = prep(si[s].wq, H_, H_); sq[s].wf = prep(si[s].wf, H_, H_);
        sq[s].wi = prep(si[s].wi, H_, H_); sq[s].wg = prep(si[s].wg, H_, H_);
        sq[s].wo = prep(si[s].wo, H_, H_);
        sq[s].mg = prep(si[s].mg, 2 * MLP_, H_); sq[s].md = prep(si[s].md, H_, MLP_);
        sq[s].pw = prep(si[s].pw, H_, 2 * H_);
    }
    QW qflin = prep(flin_w, IN_, H_);

    float* emb  = (float*)alloc(FREQ_ * 4);
    float* h1v  = (float*)alloc(H_ * 4);
    float* temb = (float*)alloc(H_ * 4);
    float* stemb= (float*)alloc(H_ * 4);
    float* hbuf = (float*)alloc((size_t)LS_ * H_ * 4);
    float* cbuf = (float*)alloc((size_t)LC_ * H_ * 4);
    float* zbuf = (float*)alloc((size_t)TT_ * H_ * 4);
    float* nb1  = (float*)alloc((size_t)TT_ * H_ * 4);
    float* nb2  = (float*)alloc((size_t)LC_ * H_ * 4);
    float* qj   = (float*)alloc((size_t)TT_ * H_ * 4);
    float* qb   = (float*)alloc((size_t)TT_ * H_ * 4);
    float* fb   = (float*)alloc((size_t)TT_ * H_ * 4);
    float* vb   = (float*)alloc((size_t)TT_ * H_ * 4);
    float* gb   = (float*)alloc((size_t)TT_ * H_ * 4);
    float* ob   = (float*)alloc((size_t)TT_ * H_ * 4);
    float* aob  = (float*)alloc((size_t)TT_ * H_ * 4);
    float* big  = (float*)alloc((size_t)TT_ * 2 * MLP_ * 4);
    float* mid  = (float*)alloc((size_t)TT_ * MLP_ * 4);
    float* catb = (float*)alloc((size_t)TT_ * 2 * H_ * 4);
    float* tmp  = (float*)alloc((size_t)TT_ * H_ * 4);
    float* e1   = (float*)alloc(6 * H_ * 4);
    float* e2   = (float*)alloc(6 * H_ * 4);
    float* e3   = (float*)alloc(2 * H_ * 4);
    int8_t* aq8 = (int8_t*)alloc((size_t)TT_ * MLP_);
    float* asc  = (float*)alloc(TT_ * 4);

    auto blks = [](int n) { return (unsigned)((n + 255) / 256); };
    auto aq = [&](const float* X, int M, int K) {
        k_act_quant<<<M, 256, 0, stream>>>(X, K, aq8, asc);
    };
    auto gemm = [&](const QW& w, const float* bias, float* C, int M, int ldc, int row0) {
        dim3 g(w.N / 64, M / 64);
        gemm_iu8<<<g, 128, 0, stream>>>(aq8 + (size_t)row0 * w.K, asc + row0,
                                        w.q, w.scale, bias, C, M, w.N, w.K, ldc);
    };

    auto attn = [&](const float* xin, int T, const QW& wq, const QW& wf, const QW& wi,
                    const QW& wg, const QW& wo, const float* gn, float* out, int oldc) {
        aq(xin, T, H_);
        gemm(wq, nullptr, qb, T, H_, 0);
        gemm(wf, nullptr, fb, T, H_, 0);
        gemm(wi, nullptr, vb, T, H_, 0);
        gemm(wg, nullptr, gb, T, H_, 0);
        k_sigmoid<<<blks(T * H_), 256, 0, stream>>>(fb, T * H_);
        k_hgrn_scan<<<NH_, 256, 0, stream>>>(qb, fb, vb, ob, T);
        k_gnorm_gate<<<T * NH_, 64, 0, stream>>>(ob, gb, gn);
        aq(ob, T, H_);
        gemm(wo, nullptr, out, T, oldc, 0);
    };
    auto mlp = [&](const float* xin, int M, const QW& wgate, const QW& wdown,
                   float* out, int oldc) {
        aq(xin, M, H_);
        gemm(wgate, nullptr, big, M, 2 * MLP_, 0);
        k_silugate<<<blks(M * MLP_), 256, 0, stream>>>(big, mid, M * MLP_);
        aq(mid, M, MLP_);
        gemm(wdown, nullptr, out, M, oldc, 0);
    };

    k_timestep_emb<<<1, 128, 0, stream>>>(t, emb);
    k_matvec<<<blks(H_), 256, 0, stream>>>(te_w1, te_b1, emb, h1v, H_, FREQ_, 1);
    k_matvec<<<blks(H_), 256, 0, stream>>>(te_w2, te_b2, h1v, temb, H_, H_, 0);
    k_silu<<<blks(H_), 256, 0, stream>>>(temb, stemb, H_);

    aq(x, LS_, IN_);
    gemm(qxemb, xemb_b, hbuf, LS_, H_, 0);
    k_copy<<<blks(LC_ * H_), 256, 0, stream>>>(cbuf, context, LC_ * H_);

    for (int d = 0; d < 2; ++d) {
        k_matvec<<<blks(6 * H_), 256, 0, stream>>>(du[d].ada_x_w, du[d].ada_x_b, stemb, e1, 6 * H_, H_, 0);
        k_matvec<<<blks(6 * H_), 256, 0, stream>>>(du[d].ada_c_w, du[d].ada_c_b, stemb, e2, 6 * H_, H_, 0);
        k_ln_mod<<<LS_, 256, 0, stream>>>(hbuf, e1, e1 + H_, nb1);
        k_ln_mod<<<LC_, 256, 0, stream>>>(cbuf, e2, e2 + H_, nb2);
        aq(nb2, LC_, H_); gemm(dq[d].add_q, nullptr, qj, LC_, H_, 0);
        aq(nb1, LS_, H_); gemm(dq[d].to_q, nullptr, qj + (size_t)LC_ * H_, LS_, H_, 0);
        attn(qj, TT_, dq[d].wq, dq[d].wf, dq[d].wi, dq[d].wg, dq[d].wo, du[d].gnorm, aob, H_);
        aq(aob, TT_, H_);
        gemm(dq[d].to_out, nullptr, tmp, LS_, H_, LC_);
        k_gated_add<<<blks(LS_ * H_), 256, 0, stream>>>(hbuf, tmp, e1 + 2 * H_, LS_ * H_);
        gemm(dq[d].to_add_out, nullptr, tmp, LC_, H_, 0);
        k_gated_add<<<blks(LC_ * H_), 256, 0, stream>>>(cbuf, tmp, e2 + 2 * H_, LC_ * H_);
        k_ln_mod<<<LS_, 256, 0, stream>>>(hbuf, e1 + 3 * H_, e1 + 4 * H_, nb1);
        mlp(nb1, LS_, dq[d].ffg, dq[d].ffd, tmp, H_);
        k_gated_add<<<blks(LS_ * H_), 256, 0, stream>>>(hbuf, tmp, e1 + 5 * H_, LS_ * H_);
        k_ln_mod<<<LC_, 256, 0, stream>>>(cbuf, e2 + 3 * H_, e2 + 4 * H_, nb2);
        mlp(nb2, LC_, dq[d].ffcg, dq[d].ffcd, tmp, H_);
        k_gated_add<<<blks(LC_ * H_), 256, 0, stream>>>(cbuf, tmp, e2 + 5 * H_, LC_ * H_);
    }

    k_copy<<<blks(LC_ * H_), 256, 0, stream>>>(zbuf, cbuf, LC_ * H_);
    k_copy<<<blks(LS_ * H_), 256, 0, stream>>>(zbuf + (size_t)LC_ * H_, hbuf, LS_ * H_);
    for (int s = 0; s < 2; ++s) {
        k_matvec<<<blks(2 * H_), 256, 0, stream>>>(si[s].ada_w, si[s].ada_b, stemb, e3, 2 * H_, H_, 0);
        k_ln_mod<<<TT_, 256, 0, stream>>>(zbuf, e3, nullptr, nb1);
        attn(nb1, TT_, sq[s].wq, sq[s].wf, sq[s].wi, sq[s].wg, sq[s].wo,
             si[s].gnorm, catb, 2 * H_);
        mlp(nb1, TT_, sq[s].mg, sq[s].md, catb + H_, 2 * H_);
        aq(catb, TT_, 2 * H_);
        gemm(sq[s].pw, si[s].pb, tmp, TT_, H_, 0);
        k_gated_add<<<blks(TT_ * H_), 256, 0, stream>>>(zbuf, tmp, e3 + H_, TT_ * H_);
    }

    k_matvec<<<blks(2 * H_), 256, 0, stream>>>(fada_w, fada_b, stemb, e3, 2 * H_, H_, 0);
    k_ln_mod<<<LS_, 256, 0, stream>>>(zbuf + (size_t)LC_ * H_, e3 + H_, e3, nb1);
    aq(nb1, LS_, H_);
    gemm(qflin, flin_b, (float*)d_out, LS_, IN_, 0);
}
